// PrototypeBank_65850438582450
// MI455X (gfx1250) — matrix operand order repaired and checked by reading, not compiled, not run on hardware
//
#include <hip/hip_runtime.h>

// ---------------------------------------------------------------------------
// PrototypeBank update for MI455X (gfx1250, wave32, WMMA).
//   sims GEMM (131072x1024x512) via v_wmma_f32_16x16x32_bf16, argmax in-reg,
//   segment-sum via fp32 global atomics, EMA+renorm finalize.
// ---------------------------------------------------------------------------

typedef __attribute__((ext_vector_type(16))) __bf16         v16bf;
typedef __attribute__((ext_vector_type(8)))  float          v8f;
typedef __attribute__((ext_vector_type(8)))  unsigned short v8us;
typedef __attribute__((ext_vector_type(16))) unsigned short v16us;

#define DDIM  512      // feature dim (reference)
#define MTILE 128      // embedding rows per block (8 waves x 16)
#define LDA   520      // padded LDS row stride in ushorts (16B aligned, bank-skewed)
#define EPS   1e-6f

__device__ __forceinline__ unsigned short f2bf(float f) {
    union { float f; unsigned int u; } c; c.f = f;
    unsigned int u = c.u;
    unsigned int r = u + 0x7FFFu + ((u >> 16) & 1u);   // round-to-nearest-even
    return (unsigned short)(r >> 16);
}

// ---------------------------------------------------------------------------
// Kernel 1: normalize prototypes, store bf16 row-major [K][DDIM]
// ---------------------------------------------------------------------------
__global__ void __launch_bounds__(256)
proto_norm_kernel(const float* __restrict__ protos,
                  unsigned short* __restrict__ protosN) {
    __shared__ float red[8];
    const int k = blockIdx.x;
    const int t = threadIdx.x;
    const float* row = protos + (size_t)k * DDIM;
    float2 x = *(const float2*)(row + t * 2);
    float ss = x.x * x.x + x.y * x.y;
    #pragma unroll
    for (int off = 16; off > 0; off >>= 1) ss += __shfl_xor(ss, off, 32);
    if ((t & 31) == 0) red[t >> 5] = ss;
    __syncthreads();
    float tot = red[0] + red[1] + red[2] + red[3] +
                red[4] + red[5] + red[6] + red[7];
    float s = 1.0f / fmaxf(sqrtf(tot), EPS);
    unsigned int packed = (unsigned int)f2bf(x.x * s) |
                          ((unsigned int)f2bf(x.y * s) << 16);
    ((unsigned int*)protosN)[(size_t)k * (DDIM / 2) + t] = packed;
}

// ---------------------------------------------------------------------------
// Kernel 2: per 128-row tile:
//   normalize rows -> bf16 LDS, WMMA sims vs all K prototypes, argmax,
//   atomic scatter of fp32 normalized embeddings into sums/counts.
// ---------------------------------------------------------------------------
__global__ void __launch_bounds__(256, 1)
sim_argmax_accum_kernel(const float* __restrict__ emb,
                        const unsigned short* __restrict__ protosN,
                        float* __restrict__ sums,
                        float* __restrict__ counts,
                        int K) {
    extern __shared__ unsigned char smemRaw[];
    unsigned short* smA    = (unsigned short*)smemRaw;              // MTILE*LDA
    float*          sInv   = (float*)(smemRaw + MTILE * LDA * 2);   // MTILE
    int*            sAsn   = (int*)(sInv + MTILE);                  // MTILE

    const int t    = threadIdx.x;
    const int wave = t >> 5;
    const int lane = t & 31;
    const int h    = lane >> 4;      // lane half (0/1)
    const int ml   = lane & 15;      // lane-in-half
    const size_t rowBase = (size_t)blockIdx.x * MTILE;

    // ---- Phase 1: normalize 128 rows into bf16 LDS ----
    for (int r = wave; r < MTILE; r += 8) {
        const float4* rp = (const float4*)(emb + (rowBase + r) * DDIM + lane * 16);
        float4 a = rp[0], b = rp[1], c4 = rp[2], d4 = rp[3];
        float ss = a.x*a.x + a.y*a.y + a.z*a.z + a.w*a.w
                 + b.x*b.x + b.y*b.y + b.z*b.z + b.w*b.w
                 + c4.x*c4.x + c4.y*c4.y + c4.z*c4.z + c4.w*c4.w
                 + d4.x*d4.x + d4.y*d4.y + d4.z*d4.z + d4.w*d4.w;
        #pragma unroll
        for (int off = 16; off > 0; off >>= 1) ss += __shfl_xor(ss, off, 32);
        float s = 1.0f / fmaxf(sqrtf(ss), EPS);
        if (lane == 0) sInv[r] = s;
        unsigned int* dst = (unsigned int*)&smA[r * LDA + lane * 16];
        dst[0] = (unsigned int)f2bf(a.x*s)  | ((unsigned int)f2bf(a.y*s)  << 16);
        dst[1] = (unsigned int)f2bf(a.z*s)  | ((unsigned int)f2bf(a.w*s)  << 16);
        dst[2] = (unsigned int)f2bf(b.x*s)  | ((unsigned int)f2bf(b.y*s)  << 16);
        dst[3] = (unsigned int)f2bf(b.z*s)  | ((unsigned int)f2bf(b.w*s)  << 16);
        dst[4] = (unsigned int)f2bf(c4.x*s) | ((unsigned int)f2bf(c4.y*s) << 16);
        dst[5] = (unsigned int)f2bf(c4.z*s) | ((unsigned int)f2bf(c4.w*s) << 16);
        dst[6] = (unsigned int)f2bf(d4.x*s) | ((unsigned int)f2bf(d4.y*s) << 16);
        dst[7] = (unsigned int)f2bf(d4.z*s) | ((unsigned int)f2bf(d4.w*s) << 16);
    }
    __syncthreads();

    // ---- Phase 2: WMMA sims + running argmax ----
    const int mBase = wave * 16;
    float best[8];
    int   bidx[8];
    #pragma unroll
    for (int v = 0; v < 8; ++v) { best[v] = -3.402823466e38f; bidx[v] = 0; }

    // A-fragment row pointer for this lane (row = mBase + ml, both halves use it)
    const unsigned short* aRow = &smA[(mBase + ml) * LDA];

    for (int nT = 0; nT < (K >> 4); ++nT) {
        // B: prototype column n = nT*16 + ml; lane half selects K offset +16h
        const unsigned short* bRow =
            protosN + (size_t)(nT * 16 + ml) * DDIM + 16 * h;

        v8f acc = {0.f, 0.f, 0.f, 0.f, 0.f, 0.f, 0.f, 0.f};
        #pragma unroll
        for (int kk = 0; kk < DDIM / 32; ++kk) {
            const int kBase = kk * 32;
            // A 16x32 bf16 layout: half h holds K = kBase+16h .. +15 (contiguous)
            union { v8us half2[2]; v16bf v; } A;
            A.half2[0] = *(const v8us*)(aRow + kBase + 16 * h);
            A.half2[1] = *(const v8us*)(aRow + kBase + 16 * h + 8);
            // B 32x16 bf16 layout: half h holds K = kBase+16h .. +15 (contiguous)
            union { v16us u; v16bf v; } B;
            B.u = *(const v16us*)(bRow + kBase);
            acc = __builtin_amdgcn_wmma_f32_16x16x32_bf16(
                false, A.v, false, B.v, (short)0, acc, false, false);
        }
        const int nGlob = nT * 16 + ml;
        #pragma unroll
        for (int v = 0; v < 8; ++v) {
            if (acc[v] > best[v]) { best[v] = acc[v]; bidx[v] = nGlob; }
        }
    }

    // Cross-lane argmax within each 16-lane half (row m = mBase + v + 8h).
    // xor masks 1..8 never cross the 16-lane halves of a wave32.
    #pragma unroll
    for (int v = 0; v < 8; ++v) {
        float bv = best[v];
        int   bi = bidx[v];
        #pragma unroll
        for (int off = 1; off < 16; off <<= 1) {
            float ov = __shfl_xor(bv, off, 32);
            int   oi = __shfl_xor(bi, off, 32);
            if (ov > bv || (ov == bv && oi < bi)) { bv = ov; bi = oi; }
        }
        if (ml == 0) sAsn[mBase + v + 8 * h] = bi;
    }
    __syncthreads();

    // ---- Phase 3: fp32 scatter accumulate (segment sum) ----
    for (int idx = t; idx < MTILE * DDIM; idx += 256) {
        int r = idx >> 9;            // /512
        int d = idx & (DDIM - 1);
        float val = emb[(rowBase + r) * DDIM + d] * sInv[r];
        atomicAdd(&sums[(size_t)sAsn[r] * DDIM + d], val);
    }
    if (t < MTILE) atomicAdd(&counts[sAsn[t]], 1.0f);
}

// ---------------------------------------------------------------------------
// Kernel 3: EMA + renormalize + select
// ---------------------------------------------------------------------------
__global__ void __launch_bounds__(256)
finalize_kernel(const float* __restrict__ protos,
                const float* __restrict__ sums,
                const float* __restrict__ counts,
                float* __restrict__ out) {
    __shared__ float red[8];
    const int k = blockIdx.x;
    const int t = threadIdx.x;
    float cnt  = counts[k];
    float invc = 1.0f / fmaxf(cnt, 1.0f);
    float2 pv = *(const float2*)(protos + (size_t)k * DDIM + t * 2);
    float2 sv = *(const float2*)(sums   + (size_t)k * DDIM + t * 2);
    float u0 = 0.9f * pv.x + 0.1f * sv.x * invc;
    float u1 = 0.9f * pv.y + 0.1f * sv.y * invc;
    float ss = u0 * u0 + u1 * u1;
    #pragma unroll
    for (int off = 16; off > 0; off >>= 1) ss += __shfl_xor(ss, off, 32);
    if ((t & 31) == 0) red[t >> 5] = ss;
    __syncthreads();
    float tot = red[0] + red[1] + red[2] + red[3] +
                red[4] + red[5] + red[6] + red[7];
    float s   = 1.0f / fmaxf(sqrtf(tot), EPS);
    bool has  = cnt > 0.0f;
    float2 o;
    o.x = has ? u0 * s : pv.x;
    o.y = has ? u1 * s : pv.y;
    *(float2*)(out + (size_t)k * DDIM + t * 2) = o;
}

// ---------------------------------------------------------------------------
extern "C" void kernel_launch(void* const* d_in, const int* in_sizes, int n_in,
                              void* d_out, int out_size, void* d_ws, size_t ws_size,
                              hipStream_t stream) {
    (void)n_in; (void)out_size; (void)ws_size;
    const float* emb    = (const float*)d_in[0];   // [N, 512] f32
    const float* protos = (const float*)d_in[1];   // [K, 512] f32
    const int N = in_sizes[0] / DDIM;              // 131072
    const int K = in_sizes[1] / DDIM;              // 1024

    // Workspace layout: sums [K*512] f32 | counts [K] f32 | protosN [K*512] bf16
    float*          sums    = (float*)d_ws;
    float*          counts  = sums + (size_t)K * DDIM;
    unsigned short* protosN = (unsigned short*)(counts + K);

    hipMemsetAsync(sums, 0, ((size_t)K * DDIM + K) * sizeof(float), stream);

    proto_norm_kernel<<<K, 256, 0, stream>>>(protos, protosN);

    const size_t shBytes = (size_t)MTILE * LDA * 2   // bf16 A tiles
                         + (size_t)MTILE * 4         // sInv
                         + (size_t)MTILE * 4;        // sAssign
    sim_argmax_accum_kernel<<<N / MTILE, 256, shBytes, stream>>>(
        emb, protosN, sums, counts, K);

    finalize_kernel<<<K, 256, 0, stream>>>(protos, sums, counts, (float*)d_out);
}
